// BiDirectionalCrossAttention_36060545417289
// MI455X (gfx1250) — matrix operand order repaired and checked by reading, not compiled, not run on hardware
//
#include <hip/hip_runtime.h>
#include <hip/hip_bf16.h>

// ---------------------------------------------------------------------------
// BiDirectionalCrossAttention for MI455X (gfx1250, wave32, WMMA f16->f32)
// Sizes: B=4, C=512, NH=8, HD=64, NQ=1024 q-tokens/batch, NKV=2048 kv-tokens.
// ---------------------------------------------------------------------------

#define BB   4
#define CC   512
#define NH   8
#define HD   64
#define NQ   1024
#define NKV  2048
#define PLD  40   // LDS row stride (halves) for P staging; keeps 16B alignment

typedef __attribute__((ext_vector_type(16))) _Float16 v16h;
typedef __attribute__((ext_vector_type(8)))  _Float16 v8h;
typedef __attribute__((ext_vector_type(8)))  float    v8f;

// ---- WMMA wrapper (codegen-confirmed builtin) ------------------------------
__device__ __forceinline__ v8f wmma_f16(v16h a, v16h b, v8f c) {
  return __builtin_amdgcn_wmma_f32_16x16x32_f16(
      /*neg_a=*/false, a, /*neg_b=*/false, b,
      /*c_mod=*/(short)0, c, /*reuse_a=*/false, /*reuse_b=*/false);
}

// ---- Fragment loaders (contiguous k order, same for A and B, wave32) ------------------
// A (16xK row-major, ld in halves): lane L holds row M=L%16; elem i holds
// K = (L/16)*16 + i  -> two contiguous 8-half (16B) chunks.
__device__ __forceinline__ v16h load_a_frag(const _Float16* __restrict__ base,
                                            int ld, int lane, int k0) {
  int m = lane & 15, g = (lane >> 4) & 1;
  const _Float16* p = base + (size_t)m * ld + k0 + g * 16;
  v8h lo = *(const v8h*)(p);
  v8h hi = *(const v8h*)(p + 8);
  v16h r;
#pragma unroll
  for (int i = 0; i < 8; ++i) { r[i] = lo[i]; r[i + 8] = hi[i]; }
  return r;
}

// B (Kx16), supplied as Bt[N][K] row-major: lane L holds col N=L%16; elem i
// holds K = (L/16)*16 + i  -> one contiguous 16-half (32B) chunk.
__device__ __forceinline__ v16h load_b_frag(const _Float16* __restrict__ bt,
                                            int ld, int lane, int k0) {
  int n = lane & 15, g = (lane >> 4) & 1;
  return *(const v16h*)(bt + (size_t)n * ld + k0 + g * 16);
}

// C/D (16x16 f32): vgpr r, lane L -> M = (L/16)*8 + r, N = L%16.

// ---------------------------------------------------------------------------
// Shared GEMM core: one wave computes a 32(M) x 64(N) tile over K=512 with
// software-pipelined fragment loads (next k-chunk in flight under WMMAs).
// ---------------------------------------------------------------------------
__device__ __forceinline__ void gemm_core_32x64(const _Float16* __restrict__ A,
                                                const _Float16* __restrict__ Wt,
                                                int lane, v8f acc[2][4]) {
  v16h a0  = load_a_frag(A,            CC, lane, 0);
  v16h a1  = load_a_frag(A + 16 * CC,  CC, lane, 0);
  v16h bf0 = load_b_frag(Wt,               CC, lane, 0);
  v16h bf1 = load_b_frag(Wt + 16 * CC,     CC, lane, 0);
  v16h bf2 = load_b_frag(Wt + 32 * CC,     CC, lane, 0);
  v16h bf3 = load_b_frag(Wt + 48 * CC,     CC, lane, 0);
#pragma unroll
  for (int k0 = 0; k0 < CC; k0 += 32) {
    v16h na0, na1, nb0, nb1, nb2, nb3;
    const bool more = (k0 + 32) < CC;
    if (more) {
      na0 = load_a_frag(A,           CC, lane, k0 + 32);
      na1 = load_a_frag(A + 16 * CC, CC, lane, k0 + 32);
      nb0 = load_b_frag(Wt,           CC, lane, k0 + 32);
      nb1 = load_b_frag(Wt + 16 * CC, CC, lane, k0 + 32);
      nb2 = load_b_frag(Wt + 32 * CC, CC, lane, k0 + 32);
      nb3 = load_b_frag(Wt + 48 * CC, CC, lane, k0 + 32);
    }
    acc[0][0] = wmma_f16(a0, bf0, acc[0][0]);
    acc[1][0] = wmma_f16(a1, bf0, acc[1][0]);
    acc[0][1] = wmma_f16(a0, bf1, acc[0][1]);
    acc[1][1] = wmma_f16(a1, bf1, acc[1][1]);
    acc[0][2] = wmma_f16(a0, bf2, acc[0][2]);
    acc[1][2] = wmma_f16(a1, bf2, acc[1][2]);
    acc[0][3] = wmma_f16(a0, bf3, acc[0][3]);
    acc[1][3] = wmma_f16(a1, bf3, acc[1][3]);
    if (more) { a0 = na0; a1 = na1; bf0 = nb0; bf1 = nb1; bf2 = nb2; bf3 = nb3; }
  }
}

// ---------------------------------------------------------------------------
// Pack: (B, C, N) fp32 -> (B, N, C) f16 via LDS tile transpose. block (32,8).
// ---------------------------------------------------------------------------
__global__ void pack_tr_kernel(const float* __restrict__ src,
                               _Float16* __restrict__ dst, int C, int N) {
  __shared__ float tile[32][33];
  int b  = blockIdx.z;
  int n0 = blockIdx.x * 32, c0 = blockIdx.y * 32;
  const float* s = src + (size_t)b * C * N;
  _Float16*    d = dst + (size_t)b * N * C;
#pragma unroll
  for (int i = threadIdx.y; i < 32; i += 8)
    tile[i][threadIdx.x] = s[(size_t)(c0 + i) * N + n0 + threadIdx.x];
  __syncthreads();
#pragma unroll
  for (int i = threadIdx.y; i < 32; i += 8)
    d[(size_t)(n0 + i) * C + c0 + threadIdx.x] = (_Float16)tile[threadIdx.x][i];
}

__global__ void cvt_f32_f16_kernel(const float* __restrict__ src,
                                   _Float16* __restrict__ dst, int n) {
  int i = blockIdx.x * 256 + threadIdx.x;
  if (i < n) dst[i] = (_Float16)src[i];
}

// ---------------------------------------------------------------------------
// Q projection: Qh[b*8+h][nq][d] = (qf16 @ WqkvQ^T + bqkv[0:512]) * 1/8
// grid(128 Mtiles(32 rows), 8 Nstrips(64)), block 32.
// ---------------------------------------------------------------------------
__global__ __launch_bounds__(32)
void gemm_q_kernel(const _Float16* __restrict__ X,    // 4096 x 512
                   const _Float16* __restrict__ Wt,   // 512 x 512 (Bt layout)
                   const float* __restrict__ bias,    // bqkv
                   _Float16* __restrict__ Qh) {
  int lane = threadIdx.x, g = lane >> 4, nl = lane & 15;
  int mt = blockIdx.x;
  int n0 = blockIdx.y * 64;
  v8f acc[2][4] = {};
  gemm_core_32x64(X + (size_t)mt * 32 * CC, Wt + (size_t)n0 * CC, lane, acc);
#pragma unroll
  for (int half = 0; half < 2; ++half)
#pragma unroll
    for (int t = 0; t < 4; ++t) {
      int o = n0 + t * 16 + nl;
      int h = o >> 6, d = o & 63;
      float bv = bias[o];
#pragma unroll
      for (int r = 0; r < 8; ++r) {
        int m = mt * 32 + half * 16 + g * 8 + r;   // global q token
        int b = m >> 10, nq = m & 1023;
        float v = (acc[half][t][r] + bv) * 0.125f; // fold softmax scale
        Qh[(((size_t)(b * NH + h)) * NQ + nq) * HD + d] = (_Float16)v;
      }
    }
}

// ---------------------------------------------------------------------------
// KV projection: K -> Kh[bh][m][d] (token-major), V -> Vt[bh][d][m] (transposed)
// Wt points at Wqkv rows 512..1535. grid(256, 16), block 32.
// ---------------------------------------------------------------------------
__global__ __launch_bounds__(32)
void gemm_kv_kernel(const _Float16* __restrict__ X,   // 8192 x 512
                    const _Float16* __restrict__ Wt,  // 1024 x 512 (rows 512..)
                    const float* __restrict__ bias,   // bqkv (+512 offset used)
                    _Float16* __restrict__ Kh,
                    _Float16* __restrict__ Vt) {
  int lane = threadIdx.x, g = lane >> 4, nl = lane & 15;
  int mt = blockIdx.x;
  int n0 = blockIdx.y * 64;
  v8f acc[2][4] = {};
  gemm_core_32x64(X + (size_t)mt * 32 * CC, Wt + (size_t)n0 * CC, lane, acc);
#pragma unroll
  for (int half = 0; half < 2; ++half)
#pragma unroll
    for (int t = 0; t < 4; ++t) {
      int o = n0 + t * 16 + nl;               // 0..1023: [0,512)=K, [512,1024)=V
      float bv = bias[512 + o];
#pragma unroll
      for (int r = 0; r < 8; ++r) {
        int m = mt * 32 + half * 16 + g * 8 + r;  // global kv token
        int b = m >> 11, mk = m & 2047;
        float v = acc[half][t][r] + bv;
        if (o < 512) {
          int h = o >> 6, d = o & 63;
          Kh[(((size_t)(b * NH + h)) * NKV + mk) * HD + d] = (_Float16)v;
        } else {
          int o2 = o - 512;
          int h = o2 >> 6, d = o2 & 63;
          Vt[(((size_t)(b * NH + h)) * HD + d) * NKV + mk] = (_Float16)v;
        }
      }
    }
}

// ---------------------------------------------------------------------------
// Flash attention: one wave per (b,h,16-row q tile); online softmax over 2048
// keys in 32-key steps. V frags + next K frags issued early so the exp/shuffle
// VALU stretch hides their latency. grid(B*NH*64), block 32.
// ---------------------------------------------------------------------------
__global__ __launch_bounds__(32)
void attn_kernel(const _Float16* __restrict__ Qh,
                 const _Float16* __restrict__ Kh,
                 const _Float16* __restrict__ Vt,
                 _Float16* __restrict__ AO) {
  __shared__ __align__(32) _Float16 plds[16 * PLD];
  int lane = threadIdx.x, g = lane >> 4, nl = lane & 15;
  int tile = blockIdx.x;
  int qt = tile & 63;
  int bh = tile >> 6;
  int b = bh >> 3, h = bh & 7;
  const _Float16* Qbase = Qh + ((size_t)bh * NQ + qt * 16) * HD;
  const _Float16* Kbase = Kh + (size_t)bh * NKV * HD;
  const _Float16* Vbase = Vt + (size_t)bh * HD * NKV;

  // Q A-fragments for the two 32-wide d chunks (loop invariant)
  v16h qf0 = load_a_frag(Qbase, HD, lane, 0);
  v16h qf1 = load_a_frag(Qbase, HD, lane, 32);

  v8f oacc[4] = {};
  float mrow[8], lrow[8];
#pragma unroll
  for (int r = 0; r < 8; ++r) { mrow[r] = -__builtin_inff(); lrow[r] = 0.f; }

  // Preload K fragments for the first 32-key step.
  v16h kA0 = load_b_frag(Kbase,           HD, lane, 0);   // keys 0..15,  d 0..31
  v16h kA1 = load_b_frag(Kbase,           HD, lane, 32);  // keys 0..15,  d 32..63
  v16h kB0 = load_b_frag(Kbase + 16 * HD, HD, lane, 0);   // keys 16..31, d 0..31
  v16h kB1 = load_b_frag(Kbase + 16 * HD, HD, lane, 32);

  for (int m0 = 0; m0 < NKV; m0 += 32) {
    // V fragments for this step (independent of softmax) -- issue early.
    v16h vf0 = load_b_frag(Vbase,            NKV, lane, m0);
    v16h vf1 = load_b_frag(Vbase + 16 * NKV, NKV, lane, m0);
    v16h vf2 = load_b_frag(Vbase + 32 * NKV, NKV, lane, m0);
    v16h vf3 = load_b_frag(Vbase + 48 * NKV, NKV, lane, m0);

    // S tiles for keys [m0, m0+16) and [m0+16, m0+32)
    v8f s0 = {}, s1 = {};
    s0 = wmma_f16(qf0, kA0, s0);
    s0 = wmma_f16(qf1, kA1, s0);
    s1 = wmma_f16(qf0, kB0, s1);
    s1 = wmma_f16(qf1, kB1, s1);

    // Next step's K fragments in flight during softmax math.
    const bool more = (m0 + 32) < NKV;
    v16h nA0, nA1, nB0, nB1;
    if (more) {
      const _Float16* nk = Kbase + (size_t)(m0 + 32) * HD;
      nA0 = load_b_frag(nk,           HD, lane, 0);
      nA1 = load_b_frag(nk,           HD, lane, 32);
      nB0 = load_b_frag(nk + 16 * HD, HD, lane, 0);
      nB1 = load_b_frag(nk + 16 * HD, HD, lane, 32);
    }
    if (m0 + 64 < NKV)  // prefetch next-next K tile: one 128B row per lane
      __builtin_prefetch(Kbase + (size_t)(m0 + 64 + lane) * HD, 0, 1);

    // Online softmax update. Row M=g*8+r lives in the 16 lanes of half g;
    // xor masks 1,2,4,8 reduce within that half.
    float fac[8];
#pragma unroll
    for (int r = 0; r < 8; ++r) {
      float mx = fmaxf(s0[r], s1[r]);
      mx = fmaxf(mx, __shfl_xor(mx, 1, 32));
      mx = fmaxf(mx, __shfl_xor(mx, 2, 32));
      mx = fmaxf(mx, __shfl_xor(mx, 4, 32));
      mx = fmaxf(mx, __shfl_xor(mx, 8, 32));
      float nm = fmaxf(mrow[r], mx);
      float f  = __expf(mrow[r] - nm);
      float p0 = __expf(s0[r] - nm);
      float p1 = __expf(s1[r] - nm);
      s0[r] = p0; s1[r] = p1;
      float rs = p0 + p1;
      rs += __shfl_xor(rs, 1, 32);
      rs += __shfl_xor(rs, 2, 32);
      rs += __shfl_xor(rs, 4, 32);
      rs += __shfl_xor(rs, 8, 32);
      lrow[r] = lrow[r] * f + rs;
      mrow[r] = nm;
      fac[r] = f;
    }
#pragma unroll
    for (int t = 0; t < 4; ++t)
#pragma unroll
      for (int r = 0; r < 8; ++r) oacc[t][r] *= fac[r];

    // Stage P (16 q x 32 keys) through LDS: C-layout -> A-layout.
    __syncthreads();
#pragma unroll
    for (int r = 0; r < 8; ++r) {
      plds[(g * 8 + r) * PLD + nl]      = (_Float16)s0[r];
      plds[(g * 8 + r) * PLD + 16 + nl] = (_Float16)s1[r];
    }
    __syncthreads();
    v16h pf = load_a_frag(plds, PLD, lane, 0);
    oacc[0] = wmma_f16(pf, vf0, oacc[0]);
    oacc[1] = wmma_f16(pf, vf1, oacc[1]);
    oacc[2] = wmma_f16(pf, vf2, oacc[2]);
    oacc[3] = wmma_f16(pf, vf3, oacc[3]);

    if (more) { kA0 = nA0; kA1 = nA1; kB0 = nB0; kB1 = nB1; }
  }

  // Normalize and store to attn_out[b][token][h*64+d] (f16, token-major)
#pragma unroll
  for (int t = 0; t < 4; ++t)
#pragma unroll
    for (int r = 0; r < 8; ++r) {
      float v = oacc[t][r] / lrow[r];
      size_t tok = (size_t)b * NQ + qt * 16 + g * 8 + r;
      AO[tok * CC + h * HD + t * 16 + nl] = (_Float16)v;
    }
}

// ---------------------------------------------------------------------------
// Output projection + scatter back to (B, C, H, W) fp32. grid(128, 8), block 32.
// ---------------------------------------------------------------------------
__global__ __launch_bounds__(32)
void gemm_out_kernel(const _Float16* __restrict__ X,   // 4096 x 512 (attn out)
                     const _Float16* __restrict__ Wt,  // Wout f16 (Bt layout)
                     const float* __restrict__ bias,   // bout
                     float* __restrict__ out) {
  int lane = threadIdx.x, g = lane >> 4, nl = lane & 15;
  int mt = blockIdx.x;
  int n0 = blockIdx.y * 64;
  v8f acc[2][4] = {};
  gemm_core_32x64(X + (size_t)mt * 32 * CC, Wt + (size_t)n0 * CC, lane, acc);
#pragma unroll
  for (int half = 0; half < 2; ++half)
#pragma unroll
    for (int t = 0; t < 4; ++t) {
      int o = n0 + t * 16 + nl;
      float bv = bias[o];
#pragma unroll
      for (int r = 0; r < 8; ++r) {
        int m = mt * 32 + half * 16 + g * 8 + r;   // global token
        int b = m >> 10, n = m & 1023;             // n = h*W+w
        out[((size_t)b * CC + o) * NQ + n] = acc[half][t][r] + bv;
      }
    }
}

// ---------------------------------------------------------------------------
// Host launcher
// ---------------------------------------------------------------------------
extern "C" void kernel_launch(void* const* d_in, const int* in_sizes, int n_in,
                              void* d_out, int out_size, void* d_ws, size_t ws_size,
                              hipStream_t stream) {
  (void)in_sizes; (void)n_in; (void)out_size; (void)ws_size;
  const float* q    = (const float*)d_in[0];  // (4,512,32,32)
  const float* kv   = (const float*)d_in[1];  // (4,512,64,32)
  const float* Wqkv = (const float*)d_in[2];  // (1536,512)
  const float* bqkv = (const float*)d_in[3];  // (1536)
  const float* Wout = (const float*)d_in[4];  // (512,512)
  const float* bout = (const float*)d_in[5];  // (512)
  float* out = (float*)d_out;

  char* ws = (char*)d_ws;
  size_t off = 0;
  auto alloc = [&](size_t bytes) -> void* {
    void* p = ws + off;
    off += (bytes + 255) & ~(size_t)255;
    return p;
  };
  _Float16* qf16   = (_Float16*)alloc((size_t)BB * NQ  * CC * 2);
  _Float16* kvf16  = (_Float16*)alloc((size_t)BB * NKV * CC * 2);
  _Float16* wqkv16 = (_Float16*)alloc((size_t)3 * CC * CC * 2);
  _Float16* wout16 = (_Float16*)alloc((size_t)CC * CC * 2);
  _Float16* Qh     = (_Float16*)alloc((size_t)BB * NH * NQ  * HD * 2);
  _Float16* Kh     = (_Float16*)alloc((size_t)BB * NH * NKV * HD * 2);
  _Float16* Vt     = (_Float16*)alloc((size_t)BB * NH * HD * NKV * 2);
  _Float16* AO     = (_Float16*)alloc((size_t)BB * NQ * CC * 2);

  // Pack inputs (transpose to token-major f16) and convert weights.
  pack_tr_kernel<<<dim3(NQ / 32, CC / 32, BB), dim3(32, 8), 0, stream>>>(q, qf16, CC, NQ);
  pack_tr_kernel<<<dim3(NKV / 32, CC / 32, BB), dim3(32, 8), 0, stream>>>(kv, kvf16, CC, NKV);
  {
    int n1 = 3 * CC * CC;
    cvt_f32_f16_kernel<<<(n1 + 255) / 256, 256, 0, stream>>>(Wqkv, wqkv16, n1);
    int n2 = CC * CC;
    cvt_f32_f16_kernel<<<(n2 + 255) / 256, 256, 0, stream>>>(Wout, wout16, n2);
  }

  // Projections (32-row M tiles).
  gemm_q_kernel<<<dim3(BB * NQ / 32, CC / 64), 32, 0, stream>>>(
      qf16, wqkv16, bqkv, Qh);
  gemm_kv_kernel<<<dim3(BB * NKV / 32, 2 * CC / 64), 32, 0, stream>>>(
      kvf16, wqkv16 + (size_t)CC * CC, bqkv, Kh, Vt);

  // Attention.
  attn_kernel<<<dim3(BB * NH * (NQ / 16)), 32, 0, stream>>>(Qh, Kh, Vt, AO);

  // Output projection (writes final (B,C,H,W) fp32 layout directly).
  gemm_out_kernel<<<dim3(BB * NQ / 32, CC / 64), 32, 0, stream>>>(
      AO, wout16, bout, out);
}
